// DARTSCell_19499151524329
// MI455X (gfx1250) — matrix operand order repaired and checked by reading, not compiled, not run on hardware
//
#include <hip/hip_runtime.h>
#include <math.h>

// ---------------------------------------------------------------------------
// DARTS recurrent cell for MI455X (gfx1250, wave32, WMMA bf16 16x16x32).
//   inputs [256,256,1024] f32, h0 [256,1024] f32,
//   W0 [2048,2048] f32, Ws [8,1024,2048] f32  ->  hiddens [256,256,1024] f32
// Strategy:
//   * weights converted once to bf16 AND pre-transposed to [N][K] so every
//     LDS tile fill (A, Bc, Bh) is a contiguous b128 copy (no transpose
//     scatter on the serial path; 40MB of weights stay L2-resident).
//   * Xp = x @ W0[:1024] precomputed time-parallel (65536x1024x2048 GEMM).
//   * 256 serial timesteps x 9 fused GEMM+gate kernels, double-buffered
//     GLOBAL_LOAD_ASYNC_TO_LDS_B128 staging (ASYNCcnt) overlapping WMMA.
// ---------------------------------------------------------------------------

typedef unsigned short u16;
typedef __attribute__((ext_vector_type(16))) __bf16 v16bf;
typedef __attribute__((ext_vector_type(8)))  float  v8f;
typedef int v4i __attribute__((vector_size(16)));   // matches builtin prototype

union Frag32 { uint4 u[2]; v16bf v; };          // 32B: one WMMA A or B operand
union Pack8  { u16 s[8]; uint4 u; };

#if defined(__gfx1250__) && __has_builtin(__builtin_amdgcn_global_load_async_to_lds_b128)
#define USE_ASYNC 1
#else
#define USE_ASYNC 0
#endif

// 16B global -> LDS copy: async (tracked by ASYNCcnt) when available.
__device__ __forceinline__ void cp16(const void* g, void* l) {
#if USE_ASYNC
  __builtin_amdgcn_global_load_async_to_lds_b128(
      (__attribute__((address_space(1))) v4i*)g,
      (__attribute__((address_space(3))) v4i*)l, 0, 0);
#else
  *(uint4*)l = *(const uint4*)g;
#endif
}

template <int N>
__device__ __forceinline__ void waitAsync() {
#if USE_ASYNC
#if __has_builtin(__builtin_amdgcn_s_wait_asynccnt)
  __builtin_amdgcn_s_wait_asynccnt(N);
#else
  asm volatile("s_wait_asynccnt %0" ::"i"(N) : "memory");
#endif
#endif
}

__device__ __forceinline__ u16 f2bf(float f) {   // round-to-nearest-even
  unsigned int u = __float_as_uint(f);
  u += 0x7FFFu + ((u >> 16) & 1u);
  return (u16)(u >> 16);
}
__device__ __forceinline__ float bf2f(u16 h) {
  return __uint_as_float(((unsigned int)h) << 16);
}
__device__ __forceinline__ float sigf(float x) { return 1.0f / (1.0f + __expf(-x)); }
__device__ __forceinline__ float actf(int code, float x) {
  if (code == 0) return tanhf(x);
  if (code == 1) return fmaxf(x, 0.0f);
  if (code == 2) return sigf(x);
  return x;                                      // identity
}
__device__ __forceinline__ v8f wmma_bf16(const Frag32& a, const Frag32& b, v8f c) {
  return __builtin_amdgcn_wmma_f32_16x16x32_bf16(false, a.v, false, b.v,
                                                 (short)0, c, false, false);
}

#define NHID   1024
#define BATCH  256
#define TSTEPS 256

// ---------------------------------------------------------------------------
// One-time conversion kernels: f32 -> bf16, weights transposed to [N][K].
// ---------------------------------------------------------------------------
__global__ void cvt_f2bf(const float* __restrict__ s, u16* __restrict__ d, int n) {
  for (int i = blockIdx.x * blockDim.x + threadIdx.x; i < n; i += gridDim.x * blockDim.x)
    d[i] = f2bf(s[i]);
}

// W0 [2048 k][2048 n] -> wxT [2048 n][1024 k] (k<1024), whT [2048 n][1024 k]
__global__ void cvt_w0_t(const float* __restrict__ W0, u16* __restrict__ wxT,
                         u16* __restrict__ whT) {
  const int n = 2048 * 2048;
  for (int i = blockIdx.x * blockDim.x + threadIdx.x; i < n; i += gridDim.x * blockDim.x) {
    int k = i >> 11, c = i & 2047;
    u16 v = f2bf(W0[i]);
    if (k < 1024) wxT[(size_t)c * 1024 + k] = v;
    else          whT[(size_t)c * 1024 + (k - 1024)] = v;
  }
}

// Ws [8][1024 k][2048 n] -> WsT [8][2048 n][1024 k]
__global__ void cvt_ws_t(const float* __restrict__ Ws, u16* __restrict__ WsT) {
  const int n = 8 * 1024 * 2048;
  for (int i = blockIdx.x * blockDim.x + threadIdx.x; i < n; i += gridDim.x * blockDim.x) {
    int wi = i >> 21, k = (i >> 11) & 1023, c = i & 2047;
    WsT[(size_t)wi * 2048 * 1024 + (size_t)c * 1024 + k] = f2bf(Ws[i]);
  }
}

// ---------------------------------------------------------------------------
// Xp = X @ W0x : [65536,1024] x [1024,2048] -> bf16 [65536,2048]
// 64x128 block tile, 8 waves each owning a 32x32 quadrant (2x2 WMMA frags).
// B tiles async-copied from the pre-transposed W0xT; A converted f32->bf16.
// ---------------------------------------------------------------------------
__global__ __launch_bounds__(256) void xproj_kernel(const float* __restrict__ X,
                                                    const u16* __restrict__ W0xT,
                                                    u16* __restrict__ Xp) {
  __shared__ u16 As[2][64][32];     // [m][k]
  __shared__ u16 BT[2][128][32];    // [n][k]

  const int tid  = threadIdx.x;
  const int lane = tid & 31;
  const int w    = tid >> 5;
  const int qm   = (w >> 2) * 32;          // 0 | 32
  const int qn   = (w & 3)  * 32;          // 0..96
  const int m0   = blockIdx.y * 64;
  const int n0   = blockIdx.x * 128;

  const v8f zero = {0, 0, 0, 0, 0, 0, 0, 0};
  v8f acc[2][2] = {{zero, zero}, {zero, zero}};

  const int arow = tid >> 2;               // 0..63
  const int aseg = (tid & 3) * 8;          // 0,8,16,24

  auto stageA = [&](int k0, int buf) {     // sync load + convert (f32 source)
    const float* xs = X + (size_t)(m0 + arow) * 1024 + k0 + aseg;
    float4 f0 = *(const float4*)xs;
    float4 f1 = *(const float4*)(xs + 4);
    Pack8 pk;
    pk.s[0] = f2bf(f0.x); pk.s[1] = f2bf(f0.y); pk.s[2] = f2bf(f0.z); pk.s[3] = f2bf(f0.w);
    pk.s[4] = f2bf(f1.x); pk.s[5] = f2bf(f1.y); pk.s[6] = f2bf(f1.z); pk.s[7] = f2bf(f1.w);
    *(uint4*)&As[buf][arow][aseg] = pk.u;
  };
  auto stageB = [&](int k0, int buf) {     // 2 async b128 per thread
#pragma unroll
    for (int j = 0; j < 2; ++j) {
      int u = tid * 2 + j;
      int r = u >> 2, sg = (u & 3) * 8;
      cp16(W0xT + (size_t)(n0 + r) * 1024 + k0 + sg, &BT[buf][r][sg]);
    }
  };

  stageB(0, 0);
  stageA(0, 0);
#pragma unroll 2
  for (int it = 0; it < 32; ++it) {
    const int k0  = it * 32;
    const int cur = it & 1;
    if (it + 1 < 32) {
      stageB(k0 + 32, cur ^ 1);
      waitAsync<2>();
      __builtin_prefetch((const void*)(X + (size_t)(m0 + arow) * 1024 + k0 + 32 + aseg), 0, 1);
      stageA(k0 + 32, cur ^ 1);
    } else {
      waitAsync<0>();
    }
    __syncthreads();

    Frag32 a0, a1, b0, b1;
    const int mr   = qm + (lane & 15);
    const int kh8  = (lane >> 4) * 8;      // A: lanes16-31 hold K 8..15 / 24..31
    a0.u[0] = *(const uint4*)&As[cur][mr][kh8];
    a0.u[1] = *(const uint4*)&As[cur][mr][16 + kh8];
    a1.u[0] = *(const uint4*)&As[cur][mr + 16][kh8];
    a1.u[1] = *(const uint4*)&As[cur][mr + 16][16 + kh8];
    const int nc   = qn + (lane & 15);
    const int kh16 = (lane >> 4) * 16;     // unused: B reads use kh8 (lanes16-31 hold K 8..15 / 24..31)
    b0.u[0] = *(const uint4*)&BT[cur][nc][kh8];
    b0.u[1] = *(const uint4*)&BT[cur][nc][16 + kh8];
    b1.u[0] = *(const uint4*)&BT[cur][nc + 16][kh8];
    b1.u[1] = *(const uint4*)&BT[cur][nc + 16][16 + kh8];

    acc[0][0] = wmma_bf16(a0, b0, acc[0][0]);
    acc[0][1] = wmma_bf16(a0, b1, acc[0][1]);
    acc[1][0] = wmma_bf16(a1, b0, acc[1][0]);
    acc[1][1] = wmma_bf16(a1, b1, acc[1][1]);
    __syncthreads();
  }

  const int lrow = (lane >> 4) * 8;        // C/D layout: lanes16-31 -> M+8
  const int lcol = lane & 15;
#pragma unroll
  for (int mi = 0; mi < 2; ++mi)
#pragma unroll
    for (int ni = 0; ni < 2; ++ni) {
      const int lm = qm + mi * 16 + lrow;
      const int ln = qn + ni * 16 + lcol;
#pragma unroll
      for (int r = 0; r < 8; ++r)
        Xp[(size_t)(m0 + lm + r) * 2048 + n0 + ln] = f2bf(acc[mi][ni][r]);
    }
}

// ---------------------------------------------------------------------------
// One fused DARTS node: [c|h] = Apred @ W (+ Xp), s = sp + sig(c)*(act(h)-sp).
// WT is the pre-transposed weight [2048 n][1024 k]; waves 0-3 compute the c
// columns (WT rows n0..n0+63), waves 4-7 the h columns (WT rows 1024+n0..),
// sharing the A tile. h is staged via LDS; the gate is applied in-register.
// acc_mode: 0=none, 1=Acc=s, 2=Acc+=s, 3=finalize (out=(Acc+s)/8, Hnext=bf16)
// ---------------------------------------------------------------------------
__global__ __launch_bounds__(256) void darts_step(const u16* __restrict__ Apred,
                                                  const u16* __restrict__ WT,
                                                  const u16* __restrict__ Xp,
                                                  u16* __restrict__ Sout,
                                                  float* __restrict__ Acc,
                                                  float* __restrict__ OutT,
                                                  u16* __restrict__ Hnext,
                                                  int act_code, int acc_mode) {
  __shared__ u16 As[2][64][32];
  __shared__ u16 Bc[2][64][32];
  __shared__ u16 Bh[2][64][32];
  __shared__ float hstage[64][64];

  const int tid  = threadIdx.x;
  const int lane = tid & 31;
  const int w    = tid >> 5;
  const int isH  = w >> 2;                 // 0 = c-half waves, 1 = h-half waves
  const int q    = w & 3;
  const int qm   = (q >> 1) * 32;
  const int qn   = (q & 1)  * 32;
  const int m0   = blockIdx.y * 64;
  const int n0   = blockIdx.x * 64;

  const v8f zero = {0, 0, 0, 0, 0, 0, 0, 0};
  v8f acc[2][2] = {{zero, zero}, {zero, zero}};

  const int arow = tid >> 2;               // 0..63
  const int aseg = (tid & 3) * 8;          // 0,8,16,24  (same map for A/Bc/Bh)

  auto stage = [&](int k0, int buf) {      // 3 async b128 per thread
    cp16(Apred + (size_t)(m0 + arow) * 1024 + k0 + aseg, &As[buf][arow][aseg]);
    cp16(WT + (size_t)(n0 + arow) * 1024 + k0 + aseg, &Bc[buf][arow][aseg]);
    cp16(WT + (size_t)(1024 + n0 + arow) * 1024 + k0 + aseg, &Bh[buf][arow][aseg]);
  };

  stage(0, 0);
#pragma unroll 2
  for (int it = 0; it < 32; ++it) {
    const int k0  = it * 32;
    const int cur = it & 1;
    if (it + 1 < 32) { stage(k0 + 32, cur ^ 1); waitAsync<3>(); }
    else             { waitAsync<0>(); }
    __syncthreads();

    Frag32 a0, a1, b0, b1;
    const int mr   = qm + (lane & 15);
    const int kh8  = (lane >> 4) * 8;
    a0.u[0] = *(const uint4*)&As[cur][mr][kh8];
    a0.u[1] = *(const uint4*)&As[cur][mr][16 + kh8];
    a1.u[0] = *(const uint4*)&As[cur][mr + 16][kh8];
    a1.u[1] = *(const uint4*)&As[cur][mr + 16][16 + kh8];
    const int nc   = qn + (lane & 15);
    const int kh16 = (lane >> 4) * 16;
    const u16 (*BTb)[32] = isH ? Bh[cur] : Bc[cur];
    b0.u[0] = *(const uint4*)&BTb[nc][kh8];
    b0.u[1] = *(const uint4*)&BTb[nc][16 + kh8];
    b1.u[0] = *(const uint4*)&BTb[nc + 16][kh8];
    b1.u[1] = *(const uint4*)&BTb[nc + 16][16 + kh8];

    acc[0][0] = wmma_bf16(a0, b0, acc[0][0]);
    acc[0][1] = wmma_bf16(a0, b1, acc[0][1]);
    acc[1][0] = wmma_bf16(a1, b0, acc[1][0]);
    acc[1][1] = wmma_bf16(a1, b1, acc[1][1]);
    __syncthreads();
  }

  const int lrow = (lane >> 4) * 8;
  const int lcol = lane & 15;
  if (isH) {                               // stage h + optional Xp h-half
#pragma unroll
    for (int mi = 0; mi < 2; ++mi)
#pragma unroll
      for (int ni = 0; ni < 2; ++ni) {
        const int lm = qm + mi * 16 + lrow;
        const int ln = qn + ni * 16 + lcol;
#pragma unroll
        for (int r = 0; r < 8; ++r) {
          float v = acc[mi][ni][r];
          if (Xp) v += bf2f(Xp[(size_t)(m0 + lm + r) * 2048 + (n0 + ln) + 1024]);
          hstage[lm + r][ln] = v;
        }
      }
  }
  __syncthreads();
  if (!isH) {                              // gate + state update + mean accum
#pragma unroll
    for (int mi = 0; mi < 2; ++mi)
#pragma unroll
      for (int ni = 0; ni < 2; ++ni) {
        const int lm = qm + mi * 16 + lrow;
        const int ln = qn + ni * 16 + lcol;
#pragma unroll
        for (int r = 0; r < 8; ++r) {
          const int gm = m0 + lm + r;
          const int gn = n0 + ln;
          float cc = acc[mi][ni][r];
          if (Xp) cc += bf2f(Xp[(size_t)gm * 2048 + gn]);
          const float hh = hstage[lm + r][ln];
          const float sp = bf2f(Apred[(size_t)gm * 1024 + gn]);
          const float s  = sp + sigf(cc) * (actf(act_code, hh) - sp);
          const size_t oi = (size_t)gm * 1024 + gn;
          Sout[oi] = f2bf(s);
          if (acc_mode == 1)      Acc[oi] = s;
          else if (acc_mode == 2) Acc[oi] += s;
          else if (acc_mode == 3) {
            const float hm = (Acc[oi] + s) * 0.125f;   // mean of states 1..8
            OutT[oi]  = hm;
            Hnext[oi] = f2bf(hm);
          }
        }
      }
  }
}

// ---------------------------------------------------------------------------
// Host: workspace layout (all offsets 1MB-aligned)
//   [0,4M)    W0xT bf16 [2048][1024]   [4M,8M)  W0hT bf16   [8M,40M) WsT x8
//   [40M,..)  h bf16 (512K), states[9] bf16 (4.5M)
//   [45M,46M) mean accumulator f32     [46M,302M) Xp bf16 [65536,2048]
// ---------------------------------------------------------------------------
extern "C" void kernel_launch(void* const* d_in, const int* in_sizes, int n_in,
                              void* d_out, int out_size, void* d_ws, size_t ws_size,
                              hipStream_t stream) {
  const float* X   = (const float*)d_in[0];   // [256,256,1024]
  const float* h0  = (const float*)d_in[1];   // [256,1024]
  const float* W0f = (const float*)d_in[2];   // [2048,2048]
  const float* Wsf = (const float*)d_in[3];   // [8,1024,2048]
  float* out = (float*)d_out;

  char* ws = (char*)d_ws;
  const size_t MB = 1024ull * 1024ull;
  u16*   W0xT = (u16*)(ws + 0 * MB);
  u16*   W0hT = (u16*)(ws + 4 * MB);
  u16*   WsT  = (u16*)(ws + 8 * MB);
  u16*   Hbf  = (u16*)(ws + 40 * MB);
  u16*   St   = (u16*)(ws + 40 * MB + 512 * 1024);   // 9 state buffers, 512KB each
  float* Acc  = (float*)(ws + 45 * MB);
  u16*   Xp   = (u16*)(ws + 46 * MB);

  // 1) convert + transpose weights to bf16 [N][K] (then L2-resident: 40MB << 192MB)
  cvt_w0_t<<<4096, 256, 0, stream>>>(W0f, W0xT, W0hT);
  cvt_ws_t<<<8192, 256, 0, stream>>>(Wsf, WsT);
  cvt_f2bf<<<1024, 256, 0, stream>>>(h0, Hbf, BATCH * NHID);

  // 2) time-parallel input projection: Xp = X @ W0x  (65536 x 1024 x 2048)
  xproj_kernel<<<dim3(2048 / 128, (TSTEPS * BATCH) / 64), 256, 0, stream>>>(X, W0xT, Xp);

  // 3) serial recurrence: 9 fused GEMM+gate kernels per timestep
  static const int actc[8] = {2, 1, 1, 3, 0, 2, 0, 1};  // sig,relu,relu,id,tanh,sig,tanh,relu
  static const int pred[8] = {0, 1, 1, 1, 2, 5, 3, 5};
  const size_t SSZ = (size_t)BATCH * NHID;              // elems per state buffer

  for (int t = 0; t < TSTEPS; ++t) {
    const u16* xp_t = Xp + (size_t)t * BATCH * 2048;
    // s0 = h_prev + sig(c0) * (tanh(h0) - h_prev),  [c0|h0] = Xp[t] + h @ W0h
    darts_step<<<dim3(NHID / 64, BATCH / 64), 256, 0, stream>>>(
        Hbf, W0hT, xp_t, St, nullptr, nullptr, nullptr, /*tanh*/ 0, /*acc*/ 0);
    for (int i = 0; i < 8; ++i) {
      const u16* sin  = St + (size_t)pred[i] * SSZ;
      u16*       sout = St + (size_t)(i + 1) * SSZ;
      const int  mode = (i == 0) ? 1 : ((i == 7) ? 3 : 2);
      darts_step<<<dim3(NHID / 64, BATCH / 64), 256, 0, stream>>>(
          sin, WsT + (size_t)i * 2048 * 1024, nullptr, sout, Acc,
          (i == 7) ? out + (size_t)t * SSZ : nullptr,
          (i == 7) ? Hbf : nullptr, actc[i], mode);
    }
  }
}
